// PoincareEdgeGenerator_54185307407134
// MI455X (gfx1250) — matrix operand order repaired and checked by reading, not compiled, not run on hardware
//
#include <hip/hip_runtime.h>
#include <cmath>

typedef __attribute__((ext_vector_type(16))) _Float16 v16h;
typedef __attribute__((ext_vector_type(8)))  float    v8f;
typedef __attribute__((ext_vector_type(4)))  float    f4;

#define TILE_M 128           // block tile rows
#define TILE_N 128           // block tile cols
#define KT     64            // K tile staged in LDS per iteration
#define LDSK   72            // KT + 8 halves padding -> conflict-free 16B fragment reads

// ---------------------------------------------------------------------------
// Pre-pass: x2[i] = sum_k E[i][k]^2, full F32 precision. One wave32 per row.
// ---------------------------------------------------------------------------
__global__ __launch_bounds__(256) void poincare_x2_kernel(
    const float* __restrict__ emb, float* __restrict__ x2, int n, int d)
{
    const int gwave = (blockIdx.x * blockDim.x + threadIdx.x) >> 5;
    const int lane  = threadIdx.x & 31;
    if (gwave >= n) return;
    const float* row = emb + (size_t)gwave * d;
    float s = 0.f;
    for (int k = lane; k < d; k += 32) {
        float t = row[k];
        s = fmaf(t, t, s);
    }
#pragma unroll
    for (int off = 16; off; off >>= 1)
        s += __shfl_xor(s, off, 32);
    if (lane == 0) x2[gwave] = s;
}

// ---------------------------------------------------------------------------
// Main: dot = E@E^T via v_wmma_f32_16x16x32_f16, fused Poincare epilogue.
// Block = 256 threads = 8 waves (2x4), wave tile = 64x32 (4x2 WMMA subtiles).
// launch_bounds(...,1): full VGPR budget so the 64 accumulators never spill.
// ---------------------------------------------------------------------------
__global__ __launch_bounds__(256, 1) void poincare_edge_kernel(
    const float* __restrict__ emb, const float* __restrict__ x2,
    float* __restrict__ probs, float* __restrict__ dists, int n, int d)
{
    __shared__ _Float16 ldsA[TILE_M * LDSK];
    __shared__ _Float16 ldsB[TILE_N * LDSK];

    const int tid   = threadIdx.x;
    const int lane  = tid & 31;
    const int wave  = tid >> 5;
    const int waveM = wave >> 2;          // 0..1  -> 64-row slab
    const int waveN = wave & 3;           // 0..3  -> 32-col slab
    const int rowBase = blockIdx.y * TILE_M;
    const int colBase = blockIdx.x * TILE_N;

    const int m  = lane & 15;             // row/col-within-16 per WMMA layout
    const int hi = lane >> 4;             // half-wave selector

    v8f c[4][2];
#pragma unroll
    for (int mi = 0; mi < 4; ++mi)
#pragma unroll
        for (int nj = 0; nj < 2; ++nj) {
            v8f z = {0.f, 0.f, 0.f, 0.f, 0.f, 0.f, 0.f, 0.f};
            c[mi][nj] = z;
        }

    for (int k0 = 0; k0 < d; k0 += KT) {
        // ---- cooperative stage: global f32 -> f16 in LDS (128x64 A, 128x64 B)
        // A strip: TILE_M*KT/4 = 2048 float4 chunks, 16 chunks per row.
#pragma unroll 4
        for (int t = 0; t < (TILE_M * KT / 4) / 256; ++t) {
            int id = t * 256 + tid;
            int r  = id >> 4;
            int c4 = id & 15;
            f4 v = *(const f4*)(emb + (size_t)(rowBase + r) * d + k0 + c4 * 4);
            _Float16* dst = &ldsA[r * LDSK + c4 * 4];
            dst[0] = (_Float16)v[0]; dst[1] = (_Float16)v[1];
            dst[2] = (_Float16)v[2]; dst[3] = (_Float16)v[3];
        }
#pragma unroll 4
        for (int t = 0; t < (TILE_N * KT / 4) / 256; ++t) {
            int id = t * 256 + tid;
            int r  = id >> 4;
            int c4 = id & 15;
            f4 v = *(const f4*)(emb + (size_t)(colBase + r) * d + k0 + c4 * 4);
            _Float16* dst = &ldsB[r * LDSK + c4 * 4];
            dst[0] = (_Float16)v[0]; dst[1] = (_Float16)v[1];
            dst[2] = (_Float16)v[2]; dst[3] = (_Float16)v[3];
        }
        __syncthreads();

        // ---- two 16x16x32 K-steps per staged K tile
#pragma unroll
        for (int ks = 0; ks < KT / 32; ++ks) {
            v16h a[4], b[2];
            // A fragment (16-bit 16x32): lo lanes K {0-7,16-23}, hi lanes +8
#pragma unroll
            for (int mi = 0; mi < 4; ++mi) {
                const _Float16* p =
                    &ldsA[(waveM * 64 + mi * 16 + m) * LDSK + ks * 32 + hi * 8];
#pragma unroll
                for (int t = 0; t < 8; ++t) {
                    a[mi][t]     = p[t];
                    a[mi][t + 8] = p[t + 16];
                }
            }
            // B fragment (16-bit 32x16): lo lanes K {0-7,16-23}, hi lanes +8
#pragma unroll
            for (int nj = 0; nj < 2; ++nj) {
                const _Float16* p =
                    &ldsB[(waveN * 32 + nj * 16 + m) * LDSK + ks * 32 + hi * 8];
#pragma unroll
                for (int t = 0; t < 8; ++t) {
                    b[nj][t]     = p[t];
                    b[nj][t + 8] = p[t + 16];
                }
            }
#pragma unroll
            for (int mi = 0; mi < 4; ++mi)
#pragma unroll
                for (int nj = 0; nj < 2; ++nj)
                    c[mi][nj] = __builtin_amdgcn_wmma_f32_16x16x32_f16(
                        false, a[mi], false, b[nj],
                        (short)0, c[mi][nj], false, false);
        }
        __syncthreads();
    }

    // ---- fused Poincare epilogue (C/D layout: VGPR v -> M=v (+8 for hi lanes))
    const size_t N = (size_t)n;
    float xj2v[2];
    int   jv[2];
#pragma unroll
    for (int nj = 0; nj < 2; ++nj) {
        jv[nj]   = colBase + waveN * 32 + nj * 16 + m;
        xj2v[nj] = x2[jv[nj]];
    }
#pragma unroll
    for (int mi = 0; mi < 4; ++mi) {
#pragma unroll
        for (int v = 0; v < 8; ++v) {
            const int   i   = rowBase + waveM * 64 + mi * 16 + v + 8 * hi;
            const float xi2 = x2[i];
#pragma unroll
            for (int nj = 0; nj < 2; ++nj) {
                const int   j   = jv[nj];
                const float xj2 = xj2v[nj];
                const float dot = c[mi][nj][v];
                const float A   = 1.f - 2.f * dot + xj2;
                const float Bv  = 1.f - xi2;
                const float Dv  = 1.f - 2.f * dot + xi2 * xj2;
                float num2 = A * A * xi2 - 2.f * A * Bv * dot + Bv * Bv * xj2;
                num2 = fmaxf(num2, 0.f);
                const float norm =
                    sqrtf(num2) * __builtin_amdgcn_rcpf(fmaxf(fabsf(Dv), 1e-15f));
                const float xc   = fminf(norm, 1.f - 1e-7f);      // norm >= 0
                // dist = 2*artanh(xc) = log((1+xc)/(1-xc))
                const float dist =
                    __logf((1.f + xc) * __builtin_amdgcn_rcpf(1.f - xc));
                const float prob = (i == j)
                    ? 0.f
                    : __builtin_amdgcn_rcpf(1.f + __expf(dist)); // sigmoid(-d)
                const size_t idx = (size_t)i * N + j;
                dists[idx] = dist;
                probs[idx] = prob;
            }
        }
    }
}

extern "C" void kernel_launch(void* const* d_in, const int* in_sizes, int n_in,
                              void* d_out, int out_size, void* d_ws, size_t ws_size,
                              hipStream_t stream) {
    const float* emb = (const float*)d_in[0];
    float* out = (float*)d_out;

    const long long nn = (long long)(out_size / 2);      // n*n
    const int n = (int)(std::sqrt((double)nn) + 0.5);    // 8192
    const int d = in_sizes[0] / n;                       // 256

    float* x2    = (float*)d_ws;                         // n floats of scratch
    float* probs = out;
    float* dists = out + nn;

    // x2 pre-pass: one wave per row, 8 waves per block
    dim3 gridX((unsigned)((n + 7) / 8));
    poincare_x2_kernel<<<gridX, 256, 0, stream>>>(emb, x2, n, d);

    // main WMMA kernel: 128x128 output tile per 256-thread block
    dim3 grid((unsigned)(n / TILE_N), (unsigned)(n / TILE_M));
    poincare_edge_kernel<<<grid, 256, 0, stream>>>(emb, x2, probs, dists, n, d);
}
